// StressNet2_39771397161256
// MI455X (gfx1250) — compile-verified, matrix operand order checked by reading, not run on hardware
//
#include <hip/hip_runtime.h>
#include <hip/hip_bf16.h>
#include <math.h>

// ---------------------------------------------------------------------------
// Types for CDNA5 WMMA
// ---------------------------------------------------------------------------
typedef __attribute__((ext_vector_type(16))) _Float16 v16h;
typedef __attribute__((ext_vector_type(8)))  float    v8f;

#define B_ 8
#define QE_R 64          // rows per block in the query-decoder kernel
#define LDA 520          // f16 LDS leading dims (padded)
#define LDB 264
#define LDC 136

// ---------------------------------------------------------------------------
// WMMA fragment loaders (CDNA5 16-bit 16x32 A / 32x16 B layouts, wave32).
// A is row-major [m][k] in LDS. B is pre-transposed in global: Wt[n][k]
// so the (k, k+1) pair per VGPR is contiguous (merges into b128 loads).
// ---------------------------------------------------------------------------
__device__ inline v16h ldfragA(const _Float16* A, int lda, int m0, int k0) {
  int lane = threadIdx.x & 31;
  const _Float16* p = A + (size_t)(m0 + (lane & 15)) * lda + k0 + ((lane >> 4) << 3);
  v16h a;
#pragma unroll
  for (int v = 0; v < 8; ++v) {
    int k = (v < 4) ? (2 * v) : (16 + 2 * (v - 4));
    a[2 * v]     = p[k];
    a[2 * v + 1] = p[k + 1];
  }
  return a;
}

__device__ inline v16h ldfragB(const _Float16* Wt, int K, int n0, int k0) {
  int lane = threadIdx.x & 31;
  const _Float16* p = Wt + (size_t)(n0 + (lane & 15)) * K + k0 + ((lane >> 4) << 3);
  v16h b;
#pragma unroll
  for (int v = 0; v < 8; ++v) {
    int k = (v < 4) ? (2 * v) : (16 + 2 * (v - 4));
    b[2 * v]     = p[k];
    b[2 * v + 1] = p[k + 1];
  }
  return b;
}

__device__ inline void stfragD(_Float16* Out, int ldo, int mt, int nt,
                               const v8f& acc, const float* bias) {
  int lane = threadIdx.x & 31;
  int n  = nt * 16 + (lane & 15);
  int mb = mt * 16 + ((lane >> 4) << 3);
  float bv = bias[n];
#pragma unroll
  for (int v = 0; v < 8; ++v)
    Out[(size_t)(mb + v) * ldo + n] = (_Float16)(acc[v] + bv);
}

// ---------------------------------------------------------------------------
// GEMM over a 64-row tile: Out[64xN](f16,LDS) = A[64xK](f16,LDS) * Wt^T + bias.
// Two schedules, chosen per layer:
//  * strip (nT==16, the 80%-of-FLOPs layers): each wave owns one N-tile pair
//    and all 4 M-tiles -> 8 live accumulators, B-fragments loaded once per
//    k-step (4x less weight traffic), 8 independent WMMA chains.
//  * pairs (small layers): each wave owns (M-tile, adjacent-N-tile-pair)
//    -> 2 chains sharing one A fragment; keeps all 8 waves busy.
// K multiple of 32; N/16 even for all decoder layers.
// ---------------------------------------------------------------------------
__device__ inline void qe_gemm(const _Float16* Ain, int lda,
                               const _Float16* Wt, const float* bias,
                               _Float16* Out, int ldo, int K, int N) {
  int wave = threadIdx.x >> 5;
  int nT = N >> 4;
  if (nT >= 16) {
    for (int np = wave; np < (nT >> 1); np += 8) {
      int nt0 = 2 * np, nt1 = 2 * np + 1;
      v8f acc[8];
#pragma unroll
      for (int i = 0; i < 8; ++i) { v8f z = {}; acc[i] = z; }
      for (int k0 = 0; k0 < K; k0 += 32) {
        v16h b0 = ldfragB(Wt, K, nt0 * 16, k0);
        v16h b1 = ldfragB(Wt, K, nt1 * 16, k0);
#pragma unroll
        for (int mt = 0; mt < 4; ++mt) {
          v16h a = ldfragA(Ain, lda, mt * 16, k0);
          acc[2 * mt]     = __builtin_amdgcn_wmma_f32_16x16x32_f16(
              false, a, false, b0, (short)0, acc[2 * mt], false, false);
          acc[2 * mt + 1] = __builtin_amdgcn_wmma_f32_16x16x32_f16(
              false, a, false, b1, (short)0, acc[2 * mt + 1], false, false);
        }
      }
#pragma unroll
      for (int mt = 0; mt < 4; ++mt) {
        stfragD(Out, ldo, mt, nt0, acc[2 * mt], bias);
        stfragD(Out, ldo, mt, nt1, acc[2 * mt + 1], bias);
      }
    }
  } else {
    int pairs = 4 * (nT >> 1);
    for (int t = wave; t < pairs; t += 8) {
      int mt = t & 3, np = t >> 2;
      int nt0 = 2 * np, nt1 = 2 * np + 1;
      v8f acc0 = {};
      v8f acc1 = {};
      for (int k0 = 0; k0 < K; k0 += 32) {
        v16h a  = ldfragA(Ain, lda, mt * 16, k0);
        v16h b0 = ldfragB(Wt, K, nt0 * 16, k0);
        v16h b1 = ldfragB(Wt, K, nt1 * 16, k0);
        acc0 = __builtin_amdgcn_wmma_f32_16x16x32_f16(false, a, false, b0,
                                                      (short)0, acc0, false, false);
        acc1 = __builtin_amdgcn_wmma_f32_16x16x32_f16(false, a, false, b1,
                                                      (short)0, acc1, false, false);
      }
      stfragD(Out, ldo, mt, nt0, acc0, bias);
      stfragD(Out, ldo, mt, nt1, acc1, bias);
    }
  }
}

// Per-row GroupNorm (mean/var over L features) + affine + ReLU, in place, f16.
__device__ inline void gn_relu(_Float16* X, int ld, int L,
                               const float* g, const float* b,
                               float* part, float* rowmv) {
  int tid = threadIdx.x;
  int r = tid >> 2, p = tid & 3;          // 64 rows x 4 partials
  float s = 0.f, s2 = 0.f;
  _Float16* row = X + (size_t)r * ld;
  for (int c = p; c < L; c += 4) { float v = (float)row[c]; s += v; s2 += v * v; }
  part[tid * 2] = s; part[tid * 2 + 1] = s2;
  __syncthreads();
  if (p == 0) {
    float S = 0.f, S2 = 0.f;
#pragma unroll
    for (int q = 0; q < 4; ++q) { S += part[(tid + q) * 2]; S2 += part[(tid + q) * 2 + 1]; }
    float m = S / L, v = S2 / L - m * m;
    rowmv[r * 2] = m; rowmv[r * 2 + 1] = rsqrtf(v + 1e-5f);
  }
  __syncthreads();
  int total = QE_R * L;
  for (int i = tid; i < total; i += 256) {
    int rr = i / L, c = i - rr * L;
    float val = ((float)X[(size_t)rr * ld + c] - rowmv[rr * 2]) * rowmv[rr * 2 + 1] * g[c] + b[c];
    X[(size_t)rr * ld + c] = (_Float16)fmaxf(val, 0.f);
  }
  __syncthreads();
}

// ---------------------------------------------------------------------------
// Fused query decoder: 64 rows/block, all activations in LDS, WMMA GEMMs.
// ---------------------------------------------------------------------------
__global__ __launch_bounds__(256) void k_qe(
    const float* __restrict__ query, const _Float16* __restrict__ embh,
    const _Float16* w1t, const float* b1, const float* g1, const float* be1,   // 3->64
    const _Float16* w2t, const float* b2, const float* g2, const float* be2,   // 64->128
    const _Float16* w3t, const float* b3, const float* g3, const float* be3,   // 128->256
    const _Float16* wf1, const float* bf1, const float* gf1, const float* bef1,// 512->256 (s)
    const _Float16* wf2, const float* bf2, const float* gf2, const float* bef2,// 256->128 (s)
    const float* wf3, const float* bf3,                                        // 128->1 (s)
    const _Float16* wo1, const float* bo1, const float* go1, const float* beo1,// 512->256 (o)
    const _Float16* wo2, const float* bo2, const float* go2, const float* beo2,// 256->128 (o)
    const float* wo3, const float* bo3,                                        // 128->1 (o)
    float* __restrict__ out_s, float* __restrict__ out_o) {
  __shared__ _Float16 A[QE_R * LDA];
  __shared__ _Float16 Bu[QE_R * LDB];
  __shared__ _Float16 Cb[QE_R * LDC];
  __shared__ float part[QE_R * 4 * 2];
  __shared__ float rowmv[QE_R * 2];

  const int tid = threadIdx.x;
  const int row0 = blockIdx.x * QE_R;
  const int batch = row0 >> 13;  // 8192 queries per batch, 64 | 8192

  // Block 0 warms L2 for the two big weight matrices (global_prefetch_b8).
  if (blockIdx.x == 0) {
    const size_t bytes = (size_t)256 * 512 * 2;
    for (size_t o = (size_t)tid * 128; o < bytes; o += 256 * 128) {
      __builtin_prefetch((const char*)wf1 + o, 0, 1);
      __builtin_prefetch((const char*)wo1 + o, 0, 1);
    }
  }

  // stage queries, K padded 3 -> 32
  for (int i = tid; i < QE_R * 32; i += 256) {
    int r = i >> 5, c = i & 31;
    float v = (c < 3) ? query[(size_t)(row0 + r) * 3 + c] : 0.f;
    Bu[(size_t)r * LDB + c] = (_Float16)v;
  }
  __syncthreads();

  qe_gemm(Bu, LDB, w1t, b1, A, LDA, 32, 64);  __syncthreads();
  gn_relu(A, LDA, 64, g1, be1, part, rowmv);
  qe_gemm(A, LDA, w2t, b2, Bu, LDB, 64, 128); __syncthreads();
  gn_relu(Bu, LDB, 128, g2, be2, part, rowmv);
  qe_gemm(Bu, LDB, w3t, b3, A + 256, LDA, 128, 256); __syncthreads();
  gn_relu(A + 256, LDA, 256, g3, be3, part, rowmv);

  // x = [emb(batch), h3] : fill cols 0..255 with the broadcast embedding
  for (int i = tid; i < QE_R * 256; i += 256) {
    int r = i >> 8, c = i & 255;
    A[(size_t)r * LDA + c] = embh[batch * 256 + c];
  }
  __syncthreads();

  // score branch
  qe_gemm(A, LDA, wf1, bf1, Bu, LDB, 512, 256); __syncthreads();
  gn_relu(Bu, LDB, 256, gf1, bef1, part, rowmv);
  qe_gemm(Bu, LDB, wf2, bf2, Cb, LDC, 256, 128); __syncthreads();
  gn_relu(Cb, LDC, 128, gf2, bef2, part, rowmv);
  if (tid < QE_R) {
    float acc = bf3[0];
    for (int c = 0; c < 128; ++c) acc += (float)Cb[(size_t)tid * LDC + c] * wf3[c];
    out_s[row0 + tid] = acc;
  }
  __syncthreads();

  // occupancy branch
  qe_gemm(A, LDA, wo1, bo1, Bu, LDB, 512, 256); __syncthreads();
  gn_relu(Bu, LDB, 256, go1, beo1, part, rowmv);
  qe_gemm(Bu, LDB, wo2, bo2, Cb, LDC, 256, 128); __syncthreads();
  gn_relu(Cb, LDC, 128, go2, beo2, part, rowmv);
  if (tid < QE_R) {
    float acc = bo3[0];
    for (int c = 0; c < 128; ++c) acc += (float)Cb[(size_t)tid * LDC + c] * wo3[c];
    out_o[row0 + tid] = 1.f / (1.f + expf(-acc));
  }
}

// ---------------------------------------------------------------------------
// SA-stage kernels (negligible FLOPs vs decoder; plain f32 VALU)
// ---------------------------------------------------------------------------
__global__ void k_transpose(const float* __restrict__ pc, float* xyz, float* pts) {
  int t = blockIdx.x * blockDim.x + threadIdx.x;
  if (t >= B_ * 2048) return;
  int n = t % 2048, b = t / 2048;
  const float* src = pc + (size_t)b * 5 * 2048;
  for (int c = 0; c < 5; ++c) {
    float v = src[(size_t)c * 2048 + n];
    pts[(size_t)t * 5 + c] = v;
    if (c < 3) xyz[(size_t)t * 3 + c] = v;
  }
}

// one block per point, LDS tree reduction over the N-term kernel-density sum
__global__ void k_density(const float* __restrict__ xyz, float* invd, int N, float bw) {
  int t = blockIdx.x;            // b*N + i
  int i = t % N, b = t / N;
  const float* X = xyz + (size_t)b * N * 3;
  float xi = X[i * 3], yi = X[i * 3 + 1], zi = X[i * 3 + 2];
  float inv2 = 1.f / (2.f * bw * bw);
  float sum = 0.f;
  for (int j = threadIdx.x; j < N; j += blockDim.x) {
    float dx = X[j * 3] - xi, dy = X[j * 3 + 1] - yi, dz = X[j * 3 + 2] - zi;
    sum += expf(-(dx * dx + dy * dy + dz * dz) * inv2);
  }
  __shared__ float sh[256];
  sh[threadIdx.x] = sum;
  __syncthreads();
  for (int o = 128; o > 0; o >>= 1) {
    if ((int)threadIdx.x < o) sh[threadIdx.x] += sh[threadIdx.x + o];
    __syncthreads();
  }
  if (threadIdx.x == 0) {
    float dens = sh[0] / (float)N / (2.5f * bw);
    invd[t] = 1.f / dens;
  }
}

__global__ void k_fps(const float* __restrict__ xyz, int N, int npoint, int* __restrict__ idx) {
  int b = blockIdx.x;
  __shared__ float dist[2048];
  __shared__ float bv[256];
  __shared__ int   bi[256];
  const float* X = xyz + (size_t)b * N * 3;
  for (int i = threadIdx.x; i < N; i += blockDim.x) dist[i] = 1e10f;
  __syncthreads();
  int far = 0;
  for (int it = 0; it < npoint; ++it) {
    if (threadIdx.x == 0) idx[b * npoint + it] = far;
    float cx = X[far * 3], cy = X[far * 3 + 1], cz = X[far * 3 + 2];
    float best = -1.f; int besti = 0;
    for (int i = threadIdx.x; i < N; i += blockDim.x) {
      float dx = X[i * 3] - cx, dy = X[i * 3 + 1] - cy, dz = X[i * 3 + 2] - cz;
      float d = dx * dx + dy * dy + dz * dz;
      float nd = fminf(dist[i], d);
      dist[i] = nd;
      if (nd > best) { best = nd; besti = i; }
    }
    bv[threadIdx.x] = best; bi[threadIdx.x] = besti;
    __syncthreads();
    for (int o = 128; o > 0; o >>= 1) {
      if ((int)threadIdx.x < o) {
        if (bv[threadIdx.x + o] > bv[threadIdx.x] ||
            (bv[threadIdx.x + o] == bv[threadIdx.x] && bi[threadIdx.x + o] < bi[threadIdx.x])) {
          bv[threadIdx.x] = bv[threadIdx.x + o];
          bi[threadIdx.x] = bi[threadIdx.x + o];
        }
      }
      __syncthreads();
    }
    far = bi[0];
    __syncthreads();
  }
}

__global__ void k_gather_new(const float* __restrict__ xyz, const int* __restrict__ idx,
                             float* nx, int S, int N) {
  int t = blockIdx.x * blockDim.x + threadIdx.x;
  if (t >= B_ * S) return;
  int b = t / S;
  int i = idx[t];
  const float* p = xyz + ((size_t)b * N + i) * 3;
  nx[(size_t)t * 3 + 0] = p[0];
  nx[(size_t)t * 3 + 1] = p[1];
  nx[(size_t)t * 3 + 2] = p[2];
}

// wave-parallel progressive-selection KNN: one wave32 per query point.
__global__ void k_knn(const float* __restrict__ newxyz, const float* __restrict__ xyz,
                      int S, int N, int K, int* __restrict__ knn) {
  int gw = (int)((blockIdx.x * blockDim.x + threadIdx.x) >> 5);
  int lane = threadIdx.x & 31;
  if (gw >= B_ * S) return;
  int b = gw / S, s = gw % S;
  const float* q = newxyz + ((size_t)b * S + s) * 3;
  const float* X = xyz + (size_t)b * N * 3;
  float qx = q[0], qy = q[1], qz = q[2];
  float lastd = -1e30f; int lasti = -1;
  for (int k = 0; k < K; ++k) {
    float bd = 1e30f; int bii = 0x7fffffff;
    for (int i = lane; i < N; i += 32) {
      float dx = X[i * 3] - qx, dy = X[i * 3 + 1] - qy, dz = X[i * 3 + 2] - qz;
      float d = dx * dx + dy * dy + dz * dz;
      bool gt = (d > lastd) || (d == lastd && i > lasti);
      bool lt = (d < bd) || (d == bd && i < bii);
      if (gt && lt) { bd = d; bii = i; }
    }
#pragma unroll
    for (int o = 16; o > 0; o >>= 1) {
      float od = __shfl_xor(bd, o, 32);
      int   oi = __shfl_xor(bii, o, 32);
      if (od < bd || (od == bd && oi < bii)) { bd = od; bii = oi; }
    }
    if (lane == 0) knn[((size_t)b * S + s) * K + k] = bii;
    lastd = bd; lasti = bii;
  }
}

__global__ void k_group(const float* __restrict__ xyz, const float* __restrict__ pts,
                        const float* __restrict__ newxyz, const float* __restrict__ invd,
                        const int* __restrict__ knn,
                        float* gxyz, float* npts, float* gdens,
                        int S, int N, int K, int C) {
  int t = blockIdx.x * blockDim.x + threadIdx.x;
  if (t >= B_ * S * K) return;
  int k = t % K, s = (t / K) % S, b = t / (K * S);
  size_t gi = ((size_t)b * S + s) * K + k;
  int i = knn[gi];
  const float* xp = xyz + ((size_t)b * N + i) * 3;
  const float* nx = newxyz + ((size_t)b * S + s) * 3;
  float gx = xp[0] - nx[0], gy = xp[1] - nx[1], gz = xp[2] - nx[2];
  gxyz[gi * 3 + 0] = gx; gxyz[gi * 3 + 1] = gy; gxyz[gi * 3 + 2] = gz;
  float* np = npts + gi * (size_t)(C + 3);
  np[0] = gx; np[1] = gy; np[2] = gz;
  const float* pp = pts + ((size_t)b * N + i) * C;
  for (int c = 0; c < C; ++c) np[3 + c] = pp[c];
  gdens[gi] = invd[(size_t)b * N + i];
}

__global__ void k_group_all(const float* __restrict__ xyz, const float* __restrict__ pts,
                            const float* __restrict__ invd,
                            float* gxyz, float* npts, float* gdens, int N, int C) {
  int t = blockIdx.x * blockDim.x + threadIdx.x;
  if (t >= B_ * N) return;
  size_t gi = (size_t)t;
  const float* xp = xyz + gi * 3;
  gxyz[gi * 3 + 0] = xp[0]; gxyz[gi * 3 + 1] = xp[1]; gxyz[gi * 3 + 2] = xp[2];
  float* np = npts + gi * (size_t)(C + 3);
  np[0] = xp[0]; np[1] = xp[1]; np[2] = xp[2];
  const float* pp = pts + gi * C;
  for (int c = 0; c < C; ++c) np[3 + c] = pp[c];
  gdens[gi] = invd[gi];
}

__global__ void k_linear(const float* __restrict__ X, const float* __restrict__ W,
                         const float* __restrict__ bias, float* __restrict__ Y,
                         int rows, int cin, int cout) {
  size_t i = (size_t)blockIdx.x * blockDim.x + threadIdx.x;
  if (i >= (size_t)rows * cout) return;
  int r = (int)(i / cout), co = (int)(i % cout);
  float acc = bias[co];
  const float* x = X + (size_t)r * cin;
  for (int c = 0; c < cin; ++c) acc += x[c] * W[(size_t)c * cout + co];
  Y[i] = acc;
}

__global__ void k_bn_stats(const float* __restrict__ Y, int rows, int cout, float* mv) {
  int c = blockIdx.x;
  __shared__ float sh0[256], sh1[256];
  float s = 0.f, s2 = 0.f;
  for (int r = threadIdx.x; r < rows; r += blockDim.x) {
    float v = Y[(size_t)r * cout + c]; s += v; s2 += v * v;
  }
  sh0[threadIdx.x] = s; sh1[threadIdx.x] = s2;
  __syncthreads();
  for (int o = 128; o > 0; o >>= 1) {
    if ((int)threadIdx.x < o) { sh0[threadIdx.x] += sh0[threadIdx.x + o]; sh1[threadIdx.x] += sh1[threadIdx.x + o]; }
    __syncthreads();
  }
  if (threadIdx.x == 0) {
    float m = sh0[0] / rows;
    mv[2 * c] = m;
    mv[2 * c + 1] = sh1[0] / rows - m * m;
  }
}

// act: 0=none 1=relu 2=sigmoid
__global__ void k_bn_apply(float* __restrict__ Y, const float* __restrict__ mv,
                           const float* __restrict__ g, const float* __restrict__ b,
                           int rows, int cout, int act) {
  size_t i = (size_t)blockIdx.x * blockDim.x + threadIdx.x;
  if (i >= (size_t)rows * cout) return;
  int c = (int)(i % cout);
  float x = (Y[i] - mv[2 * c]) * rsqrtf(mv[2 * c + 1] + 1e-5f) * g[c] + b[c];
  if (act == 1) x = fmaxf(x, 0.f);
  else if (act == 2) x = 1.f / (1.f + expf(-x));
  Y[i] = x;
}

__global__ void k_dens_norm(const float* __restrict__ gdens, float* __restrict__ d, int S, int K) {
  int t = blockIdx.x * blockDim.x + threadIdx.x;
  if (t >= B_ * S) return;
  size_t base = (size_t)t * K;
  float mx = -1e30f;
  for (int k = 0; k < K; ++k) mx = fmaxf(mx, gdens[base + k]);
  for (int k = 0; k < K; ++k) d[base + k] = gdens[base + k] / mx;
}

__global__ void k_mul(float* __restrict__ x, const float* __restrict__ d, size_t rows, int C) {
  size_t i = (size_t)blockIdx.x * blockDim.x + threadIdx.x;
  if (i >= rows * C) return;
  x[i] *= d[i / C];
}

// out[b,s,c,e] = sum_k x[b,s,k,c] * w[b,s,k,e]  (E=16); out flat index == thread index
__global__ void k_einsum(const float* __restrict__ x, const float* __restrict__ w,
                         float* __restrict__ out, int S, int K, int C) {
  size_t t = (size_t)blockIdx.x * blockDim.x + threadIdx.x;
  if (t >= (size_t)B_ * S * C * 16) return;
  int e = (int)(t % 16);
  int c = (int)((t / 16) % C);
  size_t bs = t / ((size_t)16 * C);
  size_t base = bs * K;
  float acc = 0.f;
  for (int k = 0; k < K; ++k)
    acc += x[(base + k) * C + c] * w[(base + k) * 16 + e];
  out[t] = acc;
}

__global__ void k_f2h(const float* __restrict__ x, _Float16* __restrict__ y, int n) {
  int t = blockIdx.x * blockDim.x + threadIdx.x;
  if (t < n) y[t] = (_Float16)x[t];
}

// W (K x N, row-major f32) -> Wt (N x Kpad, row-major f16, zero-padded K)
__global__ void k_w2ht(const float* __restrict__ W, _Float16* __restrict__ Wt,
                       int K, int Kpad, int N) {
  size_t t = (size_t)blockIdx.x * blockDim.x + threadIdx.x;
  if (t >= (size_t)N * Kpad) return;
  int n = (int)(t / Kpad), k = (int)(t % Kpad);
  Wt[t] = (k < K) ? (_Float16)W[(size_t)k * N + n] : (_Float16)0.f;
}

// ---------------------------------------------------------------------------
// Host: parameter index map (assumes jax tree_flatten: top-level insertion
// order pc, query, then params with recursively sorted dict keys).
//  2..33  : qe  (fc1 W/b, fc1_occ, fc1_query, fc2, fc2_occ, fc2_query,
//                fc3, fc3_occ, fc3_query, gn1 g/b, gn1_occ, gn1_query,
//                gn2, gn2_occ, gn2_query, gn3_query)
//  34..65 : sa1, 66..97 : sa2, 98..129 : sa3, each:
//   +0 bn_linear.g +1 .b | +2..+13 density{W,b,g,be}x3 | +14 linear.W +15 .b
//   +16..+19 mlp{W,b,g,be} | +20..+31 weight{W,b,g,be}x3
// ---------------------------------------------------------------------------
extern "C" void kernel_launch(void* const* d_in, const int* in_sizes, int n_in,
                              void* d_out, int out_size, void* d_ws, size_t ws_size,
                              hipStream_t stream) {
  (void)in_sizes; (void)n_in; (void)out_size; (void)ws_size;
  const float* pc = (const float*)d_in[0];
  const float* query = (const float*)d_in[1];
  auto P = [&](int i) -> const float* { return (const float*)d_in[i]; };

  char* ws = (char*)d_ws;
  size_t off = 0;
  auto alloc  = [&](size_t bytes) -> void* { void* p = ws + off; off += (bytes + 255) & ~(size_t)255; return p; };
  auto allocF = [&](size_t n) { return (float*)alloc(n * 4); };
  auto allocI = [&](size_t n) { return (int*)alloc(n * 4); };
  auto allocH = [&](size_t n) { return (_Float16*)alloc(n * 2); };

  // persistent region
  float* xyz0 = allocF((size_t)B_ * 2048 * 3);
  float* pts0 = allocF((size_t)B_ * 2048 * 5);
  float* xyz1 = allocF((size_t)B_ * 512 * 3);
  float* pts1 = allocF((size_t)B_ * 512 * 64);
  float* xyz2 = allocF((size_t)B_ * 128 * 3);
  float* pts2 = allocF((size_t)B_ * 128 * 128);
  float* emb  = allocF((size_t)B_ * 256);
  float* mv   = allocF(2 * 256);
  _Float16* embh = allocH((size_t)B_ * 256);
  _Float16* w1t  = allocH(64 * 32);
  _Float16* w2t  = allocH(128 * 64);
  _Float16* w3t  = allocH(256 * 128);
  _Float16* wf1t = allocH((size_t)256 * 512);
  _Float16* wo1t = allocH((size_t)256 * 512);
  _Float16* wf2t = allocH((size_t)128 * 256);
  _Float16* wo2t = allocH((size_t)128 * 256);
  const size_t scratch = off;

  const int T = 256;
  auto blk = [&](size_t n) { return (unsigned)((n + T - 1) / T); };

  k_transpose<<<blk((size_t)B_ * 2048), T, 0, stream>>>(pc, xyz0, pts0);

  auto run_sa = [&](const float* xyz, const float* pts, int N, int C,
                    int npoint, int nsample, float bw, bool gall,
                    int pb, int cout, float* oxyz, float* opts) {
    off = scratch;
    int S = gall ? 1 : npoint;
    int K = gall ? N : nsample;
    size_t R = (size_t)B_ * S * K;
    float* invd  = allocF((size_t)B_ * N);
    float* gxyz  = allocF(R * 3);
    float* npts  = allocF(R * (size_t)(C + 3));
    float* gdens = allocF(R);
    float* y     = allocF(R * cout);
    float* d0    = allocF(R);
    float* t16   = allocF(R * 16);
    float* t8    = allocF(R * 8);
    float* u8    = allocF(R * 8);
    float* u8b   = allocF(R * 8);
    float* wbuf  = allocF(R * 16);
    float* ein   = allocF((size_t)B_ * S * cout * 16);

    k_density<<<(unsigned)(B_ * N), 256, 0, stream>>>(xyz, invd, N, bw);
    if (!gall) {
      int* fidx = allocI((size_t)B_ * npoint);
      int* knn  = allocI(R);
      k_fps<<<B_, 256, 0, stream>>>(xyz, N, npoint, fidx);
      k_gather_new<<<blk((size_t)B_ * npoint), T, 0, stream>>>(xyz, fidx, oxyz, npoint, N);
      k_knn<<<blk((size_t)B_ * S * 32), T, 0, stream>>>(oxyz, xyz, S, N, K, knn);
      k_group<<<blk(R), T, 0, stream>>>(xyz, pts, oxyz, invd, knn, gxyz, npts, gdens, S, N, K, C);
    } else {
      k_group_all<<<blk(R), T, 0, stream>>>(xyz, pts, invd, gxyz, npts, gdens, N, C);
    }
    // mlp
    k_linear<<<blk(R * cout), T, 0, stream>>>(npts, P(pb + 16), P(pb + 17), y, (int)R, C + 3, cout);
    k_bn_stats<<<cout, 256, 0, stream>>>(y, (int)R, cout, mv);
    k_bn_apply<<<blk(R * cout), T, 0, stream>>>(y, mv, P(pb + 18), P(pb + 19), (int)R, cout, 1);
    // density net: 1 -> 16 -> 8 -> 1 (sigmoid)
    k_dens_norm<<<blk((size_t)B_ * S), T, 0, stream>>>(gdens, d0, S, K);
    k_linear<<<blk(R * 16), T, 0, stream>>>(d0, P(pb + 2), P(pb + 3), t16, (int)R, 1, 16);
    k_bn_stats<<<16, 256, 0, stream>>>(t16, (int)R, 16, mv);
    k_bn_apply<<<blk(R * 16), T, 0, stream>>>(t16, mv, P(pb + 4), P(pb + 5), (int)R, 16, 1);
    k_linear<<<blk(R * 8), T, 0, stream>>>(t16, P(pb + 6), P(pb + 7), t8, (int)R, 16, 8);
    k_bn_stats<<<8, 256, 0, stream>>>(t8, (int)R, 8, mv);
    k_bn_apply<<<blk(R * 8), T, 0, stream>>>(t8, mv, P(pb + 8), P(pb + 9), (int)R, 8, 1);
    k_linear<<<blk(R), T, 0, stream>>>(t8, P(pb + 10), P(pb + 11), d0, (int)R, 8, 1);
    k_bn_stats<<<1, 256, 0, stream>>>(d0, (int)R, 1, mv);
    k_bn_apply<<<blk(R), T, 0, stream>>>(d0, mv, P(pb + 12), P(pb + 13), (int)R, 1, 2);
    k_mul<<<blk(R * cout), T, 0, stream>>>(y, d0, R, cout);
    // weight net: 3 -> 8 -> 8 -> 16
    k_linear<<<blk(R * 8), T, 0, stream>>>(gxyz, P(pb + 20), P(pb + 21), u8, (int)R, 3, 8);
    k_bn_stats<<<8, 256, 0, stream>>>(u8, (int)R, 8, mv);
    k_bn_apply<<<blk(R * 8), T, 0, stream>>>(u8, mv, P(pb + 22), P(pb + 23), (int)R, 8, 1);
    k_linear<<<blk(R * 8), T, 0, stream>>>(u8, P(pb + 24), P(pb + 25), u8b, (int)R, 8, 8);
    k_bn_stats<<<8, 256, 0, stream>>>(u8b, (int)R, 8, mv);
    k_bn_apply<<<blk(R * 8), T, 0, stream>>>(u8b, mv, P(pb + 26), P(pb + 27), (int)R, 8, 1);
    k_linear<<<blk(R * 16), T, 0, stream>>>(u8b, P(pb + 28), P(pb + 29), wbuf, (int)R, 8, 16);
    k_bn_stats<<<16, 256, 0, stream>>>(wbuf, (int)R, 16, mv);
    k_bn_apply<<<blk(R * 16), T, 0, stream>>>(wbuf, mv, P(pb + 30), P(pb + 31), (int)R, 16, 1);
    // einsum + linear + bn_linear + relu
    k_einsum<<<blk((size_t)B_ * S * cout * 16), T, 0, stream>>>(y, wbuf, ein, S, K, cout);
    k_linear<<<blk((size_t)B_ * S * cout), T, 0, stream>>>(ein, P(pb + 14), P(pb + 15), opts, B_ * S, 16 * cout, cout);
    k_bn_stats<<<cout, 256, 0, stream>>>(opts, B_ * S, cout, mv);
    k_bn_apply<<<blk((size_t)B_ * S * cout), T, 0, stream>>>(opts, mv, P(pb + 0), P(pb + 1), B_ * S, cout, 1);
  };

  run_sa(xyz0, pts0, 2048, 5, 512, 32, 0.1f, false, 34, 64, xyz1, pts1);
  run_sa(xyz1, pts1, 512, 64, 128, 64, 0.2f, false, 66, 128, xyz2, pts2);
  run_sa(xyz2, pts2, 128, 128, 1, 0, 0.4f, true, 98, 256, nullptr, emb);

  // decoder prep: f16 embedding + f16-transposed weights (L2-resident)
  k_f2h<<<blk((size_t)B_ * 256), T, 0, stream>>>(emb, embh, B_ * 256);
  k_w2ht<<<blk((size_t)64 * 32),   T, 0, stream>>>(P(6),  w1t,  3,   32,  64);
  k_w2ht<<<blk((size_t)128 * 64),  T, 0, stream>>>(P(12), w2t,  64,  64,  128);
  k_w2ht<<<blk((size_t)256 * 128), T, 0, stream>>>(P(18), w3t,  128, 128, 256);
  k_w2ht<<<blk((size_t)256 * 512), T, 0, stream>>>(P(2),  wf1t, 512, 512, 256);
  k_w2ht<<<blk((size_t)256 * 512), T, 0, stream>>>(P(4),  wo1t, 512, 512, 256);
  k_w2ht<<<blk((size_t)128 * 256), T, 0, stream>>>(P(8),  wf2t, 256, 256, 128);
  k_w2ht<<<blk((size_t)128 * 256), T, 0, stream>>>(P(10), wo2t, 256, 256, 128);

  float* out_s = (float*)d_out;
  float* out_o = out_s + (size_t)B_ * 8192;
  k_qe<<<(B_ * 8192) / QE_R, 256, 0, stream>>>(
      query, embh,
      w1t, P(7),  P(24), P(25),
      w2t, P(13), P(30), P(31),
      w3t, P(19), P(32), P(33),
      wf1t, P(3), P(20), P(21),
      wf2t, P(9), P(26), P(27),
      P(14), P(15),
      wo1t, P(5), P(22), P(23),
      wo2t, P(11), P(28), P(29),
      P(16), P(17),
      out_s, out_o);
}
